// MoE_6339371729725
// MI455X (gfx1250) — matrix operand order repaired and checked by reading, not compiled, not run on hardware
//
#include <hip/hip_runtime.h>
#include <hip/hip_bf16.h>

#define D_MODEL 512
#define HIDDEN  2048
#define N_EXPERTS 8

// WMMA block-packing geometry
#define NT1 (HIDDEN / 16)   // 128 n-tiles in GEMM1
#define KB1 (D_MODEL / 32)  // 16 k-blocks in GEMM1
#define NT2 (D_MODEL / 16)  // 32 n-tiles in GEMM2
#define KB2 (HIDDEN / 32)   // 64 k-blocks in GEMM2

typedef __attribute__((ext_vector_type(16))) __bf16 v16bf;
typedef __attribute__((ext_vector_type(8)))  float  v8f;
typedef __attribute__((ext_vector_type(4)))  unsigned int u32x4;
typedef __attribute__((ext_vector_type(4)))  int  i32x4;
typedef __attribute__((ext_vector_type(8)))  int  i32x8;

// ---------------------------------------------------------------------------
// TDM: issue a 2D tensor_load_to_lds of 4 packed-weight n-tile rows, each
// 2048 bf16 (4 consecutive 1KB k-blocks), row stride = strideElems (= KB*512).
// D# fields per CDNA5 ISA 8.3/8.4. Wave-level op, tracked by TENSORcnt.
// ---------------------------------------------------------------------------
static __device__ __forceinline__
void tdm_load_b(const __bf16* gaddr, unsigned int lds_off, long strideElems) {
    unsigned long long ga = (unsigned long long)(uintptr_t)gaddr;
    u32x4 g0;
    g0[0] = 1u;                                   // count=1, user mode
    g0[1] = lds_off;                              // LDS byte offset
    g0[2] = (unsigned int)ga;                     // global_addr[31:0]
    g0[3] = ((unsigned int)(ga >> 32) & 0x01FFFFFFu) | (2u << 30);  // addr hi | type=2
    i32x8 g1;
    g1[0] = (int)(1u << 16);                      // data_size=1 (2 bytes)
    g1[1] = (int)(2048u << 16);                   // tensor_dim0[15:0]=2048 (<<16)
    g1[2] = (int)(4u << 16);                      // tensor_dim0[31:16]=0 | tensor_dim1=4
    g1[3] = (int)(2048u << 16);                   // tensor_dim1 hi=0 | tile_dim0=2048
    g1[4] = 4;                                    // tile_dim1=4, tile_dim2=0
    g1[5] = (int)(unsigned int)strideElems;       // tensor_dim0_stride[31:0]
    g1[6] = 0;                                    // stride hi | dim1_stride lo
    g1[7] = 0;
    i32x4 z4 = {};
#if __clang_major__ >= 23
    i32x8 z8 = {};
    __builtin_amdgcn_tensor_load_to_lds(g0, g1, z4, z4, z8, 0);
#else
    __builtin_amdgcn_tensor_load_to_lds(g0, g1, z4, z4, 0);
#endif
}

// ---------------------------------------------------------------------------
// Kernel: convert x (fp32) -> bf16, elementwise.
// ---------------------------------------------------------------------------
__global__ __launch_bounds__(256)
void cvt_x_kernel(const float* __restrict__ x, __bf16* __restrict__ xbf, int n) {
    int i = blockIdx.x * blockDim.x + threadIdx.x;
    int stride = gridDim.x * blockDim.x;
    for (; i < n; i += stride) xbf[i] = (__bf16)x[i];
}

// ---------------------------------------------------------------------------
// Kernel: prepack weights [E][K][N] fp32 into bf16 WMMA B-fragment blocks.
// Block (e, nt, kb): 32(K) x 16(N) tile, 512 bf16 = 1KB; lane l's 16 fragment
// elements are one contiguous 32B run: elem j = W[e][kb*32+(l>>4)*16+j][nt*16+(l&15)]
// ---------------------------------------------------------------------------
__global__ __launch_bounds__(256)
void prepack_kernel(const float* __restrict__ W, __bf16* __restrict__ out,
                    int K, int N) {
    const int KB = K >> 5;
    const int NG = N >> 7;
    const int NT = N >> 4;
    int b   = blockIdx.x;
    int e   = b / (KB * NG);
    int rem = b % (KB * NG);
    int kb  = rem / NG;
    int ng  = rem % NG;

    __shared__ __bf16 tile[32][136];

    const float* src = W + ((size_t)e * K + (size_t)kb * 32) * N + (size_t)ng * 128;
    for (int i = threadIdx.x; i < 32 * 128; i += 256) {
        int r = i >> 7, c = i & 127;
        tile[r][c] = (__bf16)src[(size_t)r * N + c];
    }
    __syncthreads();

    const int l   = threadIdx.x & 31;
    const int nt2 = threadIdx.x >> 5;
    const int ntG = ng * 8 + nt2;
    __bf16* dst = out + ((size_t)(e * NT + ntG) * KB + kb) * 512 + l * 16;
    const int krow = (l >> 4) * 16;
    const int c0   = nt2 * 16 + (l & 15);
#pragma unroll
    for (int j = 0; j < 16; ++j) dst[j] = tile[krow + j][c0];
}

// ---------------------------------------------------------------------------
// Kernel: gating (one wave per token), top-2 + softmax + atomic append.
// ---------------------------------------------------------------------------
__global__ __launch_bounds__(256)
void moe_gate_kernel(const float* __restrict__ x,
                     const float* __restrict__ Wg,
                     const float* __restrict__ bg,
                     int* __restrict__ counts,
                     int* __restrict__ lists,
                     float* __restrict__ wlist,
                     int T) {
    const int lane   = threadIdx.x & 31;
    const int wave   = (blockIdx.x * blockDim.x + threadIdx.x) >> 5;
    const int nwaves = (gridDim.x * blockDim.x) >> 5;

    for (int t = wave; t < T; t += nwaves) {
        float s[N_EXPERTS];
#pragma unroll
        for (int e = 0; e < N_EXPERTS; ++e) s[e] = 0.f;

        const float* xr = x + (size_t)t * D_MODEL;
        for (int d = lane; d < D_MODEL; d += 32) {
            float xv = xr[d];
            const float* wr = Wg + (size_t)d * N_EXPERTS;
#pragma unroll
            for (int e = 0; e < N_EXPERTS; ++e) s[e] += xv * wr[e];
        }
#pragma unroll
        for (int off = 16; off > 0; off >>= 1) {
#pragma unroll
            for (int e = 0; e < N_EXPERTS; ++e)
                s[e] += __shfl_down(s[e], off, 32);
        }

        if (lane == 0) {
#pragma unroll
            for (int e = 0; e < N_EXPERTS; ++e) s[e] += bg[e];
            int e0 = 0;
#pragma unroll
            for (int e = 1; e < N_EXPERTS; ++e) if (s[e] > s[e0]) e0 = e;
            int e1 = (e0 == 0) ? 1 : 0;
#pragma unroll
            for (int e = 0; e < N_EXPERTS; ++e)
                if (e != e0 && s[e] > s[e1]) e1 = e;
            float w1 = __expf(s[e1] - s[e0]);
            float inv = 1.f / (1.f + w1);
            float w0 = inv;
            w1 *= inv;

            int p0 = atomicAdd(&counts[e0], 1);
            lists[e0 * T + p0] = t;
            wlist[e0 * T + p0] = w0;
            int p1 = atomicAdd(&counts[e1], 1);
            lists[e1 * T + p1] = t;
            wlist[e1 * T + p1] = w1;
        }
    }
}

// ---------------------------------------------------------------------------
// GEMM1: h[slot,0:2048] = relu(x[tok] * W1[e] + b1[e]), M=128 rows/WG.
// B staged by TDM double-buffer, A gathered per 128-wide k-chunk into LDS.
// ---------------------------------------------------------------------------
__global__ __launch_bounds__(256)
void moe_gemm1_kernel(const __bf16* __restrict__ xbf,
                      const __bf16* __restrict__ W1s,
                      const float* __restrict__ b1,
                      const int* __restrict__ counts,
                      const int* __restrict__ lists,
                      __bf16* __restrict__ h,
                      int T) {
    const int tiles = T >> 7;                  // 64 row-tiles per expert
    const int e    = blockIdx.x / tiles;
    const int tile = blockIdx.x % tiles;
    const int n    = counts[e];
    const int base = tile << 7;
    if (base >= n) return;
    int off = 0;
    for (int j = 0; j < e; ++j) off += counts[j];   // expert's h-row offset

    __shared__ __align__(16) __bf16 Ab[128][136];       // 34 KB (padded)
    __shared__ __align__(16) __bf16 Bb[2][4 * 2048];    // 2 x 16 KB
    __shared__ int toks[128];

    const int tid = threadIdx.x;
    if (tid < 128) {
        int i = base + tid;
        toks[tid] = (i < n) ? lists[e * T + i] : lists[e * T];
    }
    __syncthreads();

    const int wv = tid >> 5, lane = tid & 31;
    const int row = lane & 15, half = lane >> 4, koffA = half * 16;
    const __bf16* W1sB = W1s + (size_t)e * NT1 * KB1 * 512;
    const float*  b1e  = b1 + (size_t)e * HIDDEN;
    const unsigned int ldsB0 = (unsigned int)(uintptr_t)&Bb[0][0];
    const unsigned int ldsB1 = (unsigned int)(uintptr_t)&Bb[1][0];

    for (int np = 0; np < NT1 / 4; ++np) {          // 32 column passes of 64
        const int nt0 = np * 4;
        if (wv == 0)
            tdm_load_b(W1sB + (size_t)nt0 * KB1 * 512, ldsB0, KB1 * 512);
        v8f acc[4] = {{}, {}, {}, {}};
        for (int kc = 0; kc < 4; ++kc) {            // K chunks of 128
            // stage A chunk (gathered token rows)
            for (int lin = tid; lin < 128 * 16; lin += 256) {
                int r = lin >> 4, c = lin & 15;
                ((uint4*)&Ab[r][0])[c] =
                    ((const uint4*)(xbf + (size_t)toks[r] * D_MODEL + kc * 128))[c];
            }
            if (wv == 0) {
                if (kc < 3) {
                    tdm_load_b(W1sB + ((size_t)nt0 * KB1 + (kc + 1) * 4) * 512,
                               (kc & 1) ? ldsB0 : ldsB1, KB1 * 512);
                    __builtin_amdgcn_s_wait_tensorcnt(1);
                } else {
                    __builtin_amdgcn_s_wait_tensorcnt(0);
                }
            }
            __syncthreads();
            const __bf16* Bbuf = &Bb[kc & 1][0];
#pragma unroll
            for (int kbl = 0; kbl < 4; ++kbl) {
                const int kk = kbl * 32;
                v16bf a;
#pragma unroll
                for (int i = 0; i < 8; ++i) {
                    a[i]     = Ab[wv * 16 + row][kk + koffA + i];
                    a[i + 8] = Ab[wv * 16 + row][kk + 8 + koffA + i];
                }
#pragma unroll
                for (int q = 0; q < 4; ++q) {
                    v16bf b = *(const v16bf*)(Bbuf + q * 2048 + kbl * 512 + lane * 16);
                    acc[q] = __builtin_amdgcn_wmma_f32_16x16x32_bf16(
                        false, a, false, b, (short)0, acc[q], false, false);
                }
            }
            __syncthreads();
        }
#pragma unroll
        for (int q = 0; q < 4; ++q) {
            int col = (nt0 + q) * 16 + row;
            float bias = b1e[col];
#pragma unroll
            for (int r8 = 0; r8 < 8; ++r8) {
                int rg = wv * 16 + r8 + half * 8;
                if (base + rg < n) {
                    float v = acc[q][r8] + bias;
                    h[(size_t)(off + base + rg) * HIDDEN + col] = (__bf16)(v > 0.f ? v : 0.f);
                }
            }
        }
    }
}

// ---------------------------------------------------------------------------
// GEMM2: out[tok] += w * (h[slot] * W2[e] + b2[e]), M=128 rows/WG.
// ---------------------------------------------------------------------------
__global__ __launch_bounds__(256)
void moe_gemm2_kernel(const __bf16* __restrict__ h,
                      const __bf16* __restrict__ W2s,
                      const float* __restrict__ b2,
                      const int* __restrict__ counts,
                      const int* __restrict__ lists,
                      const float* __restrict__ wlist,
                      float* __restrict__ out,
                      int T) {
    const int tiles = T >> 7;
    const int e    = blockIdx.x / tiles;
    const int tile = blockIdx.x % tiles;
    const int n    = counts[e];
    const int base = tile << 7;
    if (base >= n) return;
    int off = 0;
    for (int j = 0; j < e; ++j) off += counts[j];

    __shared__ __align__(16) __bf16 Ab[128][136];
    __shared__ __align__(16) __bf16 Bb[2][4 * 2048];
    __shared__ int   toks[128];
    __shared__ float wts[128];

    const int tid = threadIdx.x;
    if (tid < 128) {
        int i = base + tid;
        if (i < n) { toks[tid] = lists[e * T + i]; wts[tid] = wlist[e * T + i]; }
        else       { toks[tid] = lists[e * T];     wts[tid] = 0.f; }
    }
    __syncthreads();

    const int wv = tid >> 5, lane = tid & 31;
    const int row = lane & 15, half = lane >> 4, koffA = half * 16;
    const __bf16* W2sB = W2s + (size_t)e * NT2 * KB2 * 512;
    const float*  b2e  = b2 + (size_t)e * D_MODEL;
    const __bf16* hB   = h + (size_t)(off + base) * HIDDEN;
    const unsigned int ldsB0 = (unsigned int)(uintptr_t)&Bb[0][0];
    const unsigned int ldsB1 = (unsigned int)(uintptr_t)&Bb[1][0];

    for (int np = 0; np < NT2 / 4; ++np) {          // 8 column passes of 64
        const int nt0 = np * 4;
        if (wv == 0)
            tdm_load_b(W2sB + (size_t)nt0 * KB2 * 512, ldsB0, KB2 * 512);
        v8f acc[4] = {{}, {}, {}, {}};
        for (int kc = 0; kc < 16; ++kc) {           // K chunks of 128 (K=2048)
            for (int lin = tid; lin < 128 * 16; lin += 256) {
                int r = lin >> 4, c = lin & 15;
                ((uint4*)&Ab[r][0])[c] =
                    ((const uint4*)(hB + (size_t)r * HIDDEN + kc * 128))[c];
            }
            if (wv == 0) {
                if (kc < 15) {
                    tdm_load_b(W2sB + ((size_t)nt0 * KB2 + (kc + 1) * 4) * 512,
                               (kc & 1) ? ldsB0 : ldsB1, KB2 * 512);
                    __builtin_amdgcn_s_wait_tensorcnt(1);
                } else {
                    __builtin_amdgcn_s_wait_tensorcnt(0);
                }
            }
            __syncthreads();
            const __bf16* Bbuf = &Bb[kc & 1][0];
#pragma unroll
            for (int kbl = 0; kbl < 4; ++kbl) {
                const int kk = kbl * 32;
                v16bf a;
#pragma unroll
                for (int i = 0; i < 8; ++i) {
                    a[i]     = Ab[wv * 16 + row][kk + koffA + i];
                    a[i + 8] = Ab[wv * 16 + row][kk + 8 + koffA + i];
                }
#pragma unroll
                for (int q = 0; q < 4; ++q) {
                    v16bf b = *(const v16bf*)(Bbuf + q * 2048 + kbl * 512 + lane * 16);
                    acc[q] = __builtin_amdgcn_wmma_f32_16x16x32_bf16(
                        false, a, false, b, (short)0, acc[q], false, false);
                }
            }
            __syncthreads();
        }
#pragma unroll
        for (int q = 0; q < 4; ++q) {
            int col = (nt0 + q) * 16 + row;
            float bias = b2e[col];
#pragma unroll
            for (int r8 = 0; r8 < 8; ++r8) {
                int rg = wv * 16 + r8 + half * 8;
                float w = wts[rg];
                if (w != 0.f) {
                    float contrib = w * (acc[q][r8] + bias);
                    float* dst = &out[(size_t)toks[rg] * D_MODEL + col];
                    __hip_atomic_fetch_add(dst, contrib, __ATOMIC_RELAXED,
                                           __HIP_MEMORY_SCOPE_AGENT);
                }
            }
        }
    }
}

extern "C" void kernel_launch(void* const* d_in, const int* in_sizes, int n_in,
                              void* d_out, int out_size, void* d_ws, size_t ws_size,
                              hipStream_t stream) {
    const float* x  = (const float*)d_in[0];
    const float* W1 = (const float*)d_in[1];
    const float* b1 = (const float*)d_in[2];
    const float* W2 = (const float*)d_in[3];
    const float* b2 = (const float*)d_in[4];
    const float* Wg = (const float*)d_in[5];
    const float* bg = (const float*)d_in[6];
    float* out = (float*)d_out;

    const int T = in_sizes[0] / D_MODEL;   // 8192 tokens

    // Workspace: counts | lists | wlist | xbf | W1s | W2s | h (2T+128 rows)
    char* ws = (char*)d_ws;
    size_t off = 0;
    int* counts = (int*)(ws + off);     off += 256;
    int* lists  = (int*)(ws + off);     off += (size_t)N_EXPERTS * T * sizeof(int);
    float* wlist = (float*)(ws + off);  off += (size_t)N_EXPERTS * T * sizeof(float);
    __bf16* xbf = (__bf16*)(ws + off);  off += (size_t)T * D_MODEL * sizeof(__bf16);
    __bf16* W1s = (__bf16*)(ws + off);  off += (size_t)N_EXPERTS * D_MODEL * HIDDEN * sizeof(__bf16);
    __bf16* W2s = (__bf16*)(ws + off);  off += (size_t)N_EXPERTS * HIDDEN * D_MODEL * sizeof(__bf16);
    __bf16* hbuf = (__bf16*)(ws + off); off += ((size_t)2 * T + 128) * HIDDEN * sizeof(__bf16);
    (void)ws_size;

    (void)hipMemsetAsync(counts, 0, 256, stream);
    (void)hipMemsetAsync(out, 0, (size_t)out_size * sizeof(float), stream);

    cvt_x_kernel<<<1024, 256, 0, stream>>>(x, xbf, T * D_MODEL);
    prepack_kernel<<<N_EXPERTS * KB1 * (HIDDEN / 128), 256, 0, stream>>>(W1, W1s, D_MODEL, HIDDEN);
    prepack_kernel<<<N_EXPERTS * KB2 * (D_MODEL / 128), 256, 0, stream>>>(W2, W2s, HIDDEN, D_MODEL);
    moe_gate_kernel<<<64, 256, 0, stream>>>(x, Wg, bg, counts, lists, wlist, T);

    const int tiles = T >> 7;              // 64 row-tiles per expert
    moe_gemm1_kernel<<<N_EXPERTS * tiles, 256, 0, stream>>>(
        xbf, W1s, b1, counts, lists, hbuf, T);
    moe_gemm2_kernel<<<N_EXPERTS * tiles, 256, 0, stream>>>(
        hbuf, W2s, b2, counts, lists, wlist, out, T);
}
